// MPNN_16587163697203
// MI455X (gfx1250) — matrix operand order repaired and checked by reading, not compiled, not run on hardware
//
#include <hip/hip_runtime.h>
#include <hip/hip_bf16.h>

#define N_NODES  8000
#define N_EDGES  16000
#define N_GRAPHS 400

typedef __attribute__((ext_vector_type(16))) __bf16 v16bf;
typedef __attribute__((ext_vector_type(2)))  __bf16 bf16x2;
typedef __attribute__((ext_vector_type(8)))  float  v8f;

__device__ __forceinline__ float silu_f(float x){ return x / (1.0f + __expf(-x)); }

// ---------------- utility kernels ----------------
__global__ void k_zero(float* __restrict__ p, int n){
  int i = blockIdx.x*blockDim.x + threadIdx.x;
  if (i < n) p[i] = 0.0f;
}

__global__ void k_deg(const int* __restrict__ dst, float* __restrict__ deg){
  int e = blockIdx.x*blockDim.x + threadIdx.x;
  if (e < N_EDGES) unsafeAtomicAdd(&deg[dst[e]], 1.0f);
}

// deg -> 1/max(deg,1)  (turn all epilogue divides into multiplies)
__global__ void k_recip(float* __restrict__ deg){
  int i = blockIdx.x*blockDim.x + threadIdx.x;
  if (i < N_NODES) deg[i] = 1.0f / fmaxf(deg[i], 1.0f);
}

// pad x [N,9] f32 -> bf16 [N,16] (zero pad)
__global__ void k_prepx(const float* __restrict__ x, __bf16* __restrict__ xp){
  int i = blockIdx.x*blockDim.x + threadIdx.x;
  if (i < N_NODES*16){
    int n = i >> 4, c = i & 15;
    xp[i] = (__bf16)(c < 9 ? x[n*9 + c] : 0.0f);
  }
}

// edge MLP stage1: h = silu(ea[E,3] @ w1[3,128] + b1) -> bf16
__global__ void k_edge_hidden(const float* __restrict__ ea, const float* __restrict__ w1,
                              const float* __restrict__ b1, __bf16* __restrict__ he){
  int i = blockIdx.x*blockDim.x + threadIdx.x;
  if (i >= N_EDGES*128) return;
  int e = i >> 7, c = i & 127;
  float a0 = ea[e*3+0], a1 = ea[e*3+1], a2 = ea[e*3+2];
  float v = a0*w1[c] + a1*w1[128+c] + a2*w1[256+c] + b1[c];
  he[i] = (__bf16)silu_f(v);
}

// Pack w2[128, fin*128] (+bias row from b2) into WMMA B-operand layout, bf16.
// Reduction index K = k_h*FPAD + i; row k_h==128 is the bias (h==1), rest zero.
// B layout (per 32x16 k-step tile): lane L: n = nt*16 + (L&15);
//   elem j (0..15): K_local = 8*(L>>4) + j (j<8), 16 + 8*(L>>4) + (j-8) (j>=8).
template<int FSH>
__global__ void k_gen_bv(const float* __restrict__ w2, const float* __restrict__ b2,
                         __bf16* __restrict__ bv, int fin, int ksteps){
  const int FPAD = 1 << FSH;
  int idx = blockIdx.x*blockDim.x + threadIdx.x;
  if (idx >= ksteps*4096) return;
  int s  = idx >> 12;
  int nt = (idx >> 9) & 7;
  int l  = (idx >> 4) & 31;
  int j  = idx & 15;
  int K  = s*32 + ((j >> 3) << 4) + ((l >> 4) << 3) + (j & 7);
  int n  = nt*16 + (l & 15);
  int kh = K >> FSH, i = K & (FPAD - 1);
  float v = 0.0f;
  if (kh < 128)       { if (i < fin) v = w2[(size_t)kh*(size_t)(fin*128) + i*128 + n]; }
  else if (kh == 128) { if (i < fin) v = b2[i*128 + n]; }
  bv[idx] = (__bf16)v;
}

// Pack root [128,128] f32 -> B layout bf16 (4 k-steps)
__global__ void k_pack_root(const float* __restrict__ root, __bf16* __restrict__ rbv){
  int idx = blockIdx.x*blockDim.x + threadIdx.x;
  if (idx >= 4*4096) return;
  int s  = idx >> 12;
  int nt = (idx >> 9) & 7;
  int l  = (idx >> 4) & 31;
  int j  = idx & 15;
  int K  = s*32 + ((j >> 3) << 4) + ((l >> 4) << 3) + (j & 7);
  int n  = nt*16 + (l & 15);
  rbv[idx] = (__bf16)root[K*128 + n];
}

// ---------------- fused message GEMM ----------------
// msg[e,o] = sum_K A[e,K]*Bv[K,o],  A[e, k_h*FPAD+i] = h[e,k_h]*x[src(e),i]
// (k_h==128 row: h=1 -> folds edge bias b2).
// 64 edges/WG, 128 threads = 4 waves; each wave owns 2 N-tiles and 4 M-tiles.
// A sub-blocks have constant k_h -> build A as (broadcast h)*(contig x) with
// v_pk_mul_bf16. LDS rows padded by 16B so the 16 lanes of each ds_load_b128
// start at banks 4m (bank-conflict-free across the 64 banks).
template<int FSH>
__global__ __launch_bounds__(128)
void k_msg_gemm(const __bf16* __restrict__ xp, const __bf16* __restrict__ he,
                const __bf16* __restrict__ bv, const int* __restrict__ src,
                const int* __restrict__ dst, float* __restrict__ agg, int ksteps){
  const int FPAD = 1 << FSH;
  const int HS = 136;          // hx row stride (bf16): 128 + 8 pad
  const int XS = FPAD + 8;     // xs row stride (bf16): +16B pad
  __shared__ __bf16 hx[64*HS];
  __shared__ __bf16 xs[64*XS];
  __shared__ int sdst[64];

  int tid   = threadIdx.x;
  int ebase = blockIdx.x * 64;
  if (tid < 64) sdst[tid] = dst[ebase + tid];

  { // stage edge-hidden rows (contiguous global -> padded LDS rows)
    const uint4* gp = (const uint4*)(he + (size_t)ebase*128);
    for (int i = tid; i < 1024; i += 128){
      int row = i >> 4, off = i & 15;
      *(uint4*)&hx[row*HS + off*8] = gp[i];
    }
  }
  { // gather source node features -> padded LDS rows
    const int U4PR = FPAD/8;
    for (int i = tid; i < 64*U4PR; i += 128){
      int row = i / U4PR, off = i - row*U4PR;
      int sn = src[ebase + row];
      *(uint4*)&xs[row*XS + off*8] = ((const uint4*)(xp + (size_t)sn*FPAD))[off];
    }
  }
  __syncthreads();

  int wave = tid >> 5, lane = tid & 31, half = lane >> 4, mrow = lane & 15;
  int nt0  = wave*2;                    // two N-tiles per wave
  int ncol = nt0*16 + (lane & 15);
  v8f acc[2][4] = {};

  for (int ks = 0; ks < ksteps; ++ks){
    union { uint4 q[2]; v16bf v; } b0, b1;
    const uint4* bp = (const uint4*)(bv + (size_t)(ks*8 + nt0)*512 + lane*16);
    b0.q[0] = bp[0];  b0.q[1] = bp[1];
    b1.q[0] = bp[64]; b1.q[1] = bp[65];   // next N-tile: +512 bf16 = +64 uint4
    #pragma unroll
    for (int t = 0; t < 4; ++t){
      int m = t*16 + mrow;
      union { bf16x2 p[8]; v16bf v; } a2;
      if constexpr (FSH == 7){
        int kh = ks >> 2;                                  // constant per k-step
        __bf16 hb = (kh < 128) ? hx[m*HS + kh] : (__bf16)1.0f;   // kh==128 -> bias row
        bf16x2 h2 = {hb, hb};
        int i0 = ((ks & 3) << 5) + 8*half;
        union { uint4 q; bf16x2 p[4]; } x0, x1;
        x0.q = *(const uint4*)&xs[m*XS + i0];              // A elems j=0..7
        x1.q = *(const uint4*)&xs[m*XS + i0 + 16];         // A elems j=8..15
        #pragma unroll
        for (int p = 0; p < 4; ++p){
          a2.p[p]   = h2 * x0.p[p];
          a2.p[4+p] = h2 * x1.p[p];
        }
      } else {                                             // FPAD==16 (layer 1)
        int kh0 = ks*2, kh1 = kh0 + 1;
        __bf16 hb0 = (kh0 < 128) ? hx[m*HS + kh0]
                                 : ((kh0 == 128) ? (__bf16)1.0f : (__bf16)0.0f);
        __bf16 hb1 = (kh1 < 128) ? hx[m*HS + kh1]
                                 : ((kh1 == 128) ? (__bf16)1.0f : (__bf16)0.0f);
        bf16x2 h20 = {hb0, hb0}, h21 = {hb1, hb1};
        union { uint4 q; bf16x2 p[4]; } x0;                // both halves reuse same x
        x0.q = *(const uint4*)&xs[m*XS + 8*half];
        #pragma unroll
        for (int p = 0; p < 4; ++p){
          a2.p[p]   = h20 * x0.p[p];
          a2.p[4+p] = h21 * x0.p[p];
        }
      }
      acc[0][t] = __builtin_amdgcn_wmma_f32_16x16x32_bf16(false, a2.v, false, b0.v,
                                                          (short)0, acc[0][t], false, false);
      acc[1][t] = __builtin_amdgcn_wmma_f32_16x16x32_bf16(false, a2.v, false, b1.v,
                                                          (short)0, acc[1][t], false, false);
    }
  }

  #pragma unroll
  for (int nti = 0; nti < 2; ++nti){
    #pragma unroll
    for (int t = 0; t < 4; ++t){
      #pragma unroll
      for (int r = 0; r < 8; ++r){
        int m = t*16 + r + 8*half;        // C/D layout: elem r -> M = r + 8*half
        int d = sdst[m];
        unsafeAtomicAdd(&agg[(size_t)d*128 + ncol + nti*16], acc[nti][t][r]);
      }
    }
  }
}

// ---------------- node updates ----------------
// layer1: out = silu(x@root1 + agg*rdeg + bias)  (K=9, VALU)
__global__ void k_node1(const float* __restrict__ x, const float* __restrict__ root,
                        const float* __restrict__ bias, const float* __restrict__ agg,
                        const float* __restrict__ rdeg, __bf16* __restrict__ xpo){
  int i = blockIdx.x*blockDim.x + threadIdx.x;
  if (i >= N_NODES*128) return;
  int n = i >> 7, c = i & 127;
  float v = bias[c];
  #pragma unroll
  for (int k = 0; k < 9; ++k) v += x[n*9 + k] * root[k*128 + c];
  v += agg[i] * rdeg[n];
  xpo[i] = (__bf16)silu_f(v);
}

// layers 2/3: WMMA x@root, fused epilogue. 32 nodes/WG, 8 waves = 8 N-tiles.
__global__ __launch_bounds__(256)
void k_node_wmma(const __bf16* __restrict__ xpi, const __bf16* __restrict__ rbv,
                 const float* __restrict__ bias, const float* __restrict__ agg,
                 const float* __restrict__ rdeg, __bf16* __restrict__ xpo){
  int tid = threadIdx.x;
  int nbase = blockIdx.x * 32;
  int wave = tid >> 5, lane = tid & 31, half = lane >> 4, mrow = lane & 15;
  int ncol = wave*16 + (lane & 15);
  v8f acc[2] = {};

  for (int s = 0; s < 4; ++s){
    union { uint4 q[2]; v16bf v; } bu;
    const uint4* bp = (const uint4*)(rbv + (size_t)(s*8 + wave)*512 + lane*16);
    bu.q[0] = bp[0]; bu.q[1] = bp[1];
    #pragma unroll
    for (int t = 0; t < 2; ++t){
      int node = nbase + t*16 + mrow;
      union { uint4 q[2]; v16bf v; } au;  // A layout: K = s*32 + 8h+j / 16+8h+(j-8)
      const __bf16* ap = xpi + (size_t)node*128 + s*32 + 8*half;
      au.q[0] = *((const uint4*)ap);
      au.q[1] = *((const uint4*)(ap + 16));
      acc[t] = __builtin_amdgcn_wmma_f32_16x16x32_bf16(false, au.v, false, bu.v,
                                                       (short)0, acc[t], false, false);
    }
  }
  #pragma unroll
  for (int t = 0; t < 2; ++t){
    #pragma unroll
    for (int r = 0; r < 8; ++r){
      int n = nbase + t*16 + r + 8*half;
      float v = acc[t][r] + agg[(size_t)n*128 + ncol] * rdeg[n] + bias[ncol];
      xpo[(size_t)n*128 + ncol] = (__bf16)silu_f(v);
    }
  }
}

// ---------------- pooling & head ----------------
__global__ void k_pool(const __bf16* __restrict__ h, const int* __restrict__ batch,
                       float* __restrict__ pooled, float* __restrict__ pcnt){
  int i = blockIdx.x*blockDim.x + threadIdx.x;
  if (i >= N_NODES*128) return;
  int n = i >> 7, c = i & 127;
  int b = batch[n];
  unsafeAtomicAdd(&pooled[b*128 + c], (float)h[i]);
  if (c == 0) unsafeAtomicAdd(&pcnt[b], 1.0f);
}

__global__ void k_head(const float* __restrict__ pooled, const float* __restrict__ pcnt,
                       const float* __restrict__ gf,
                       const float* __restrict__ gp_w, const float* __restrict__ gp_b,
                       const float* __restrict__ gbn_g, const float* __restrict__ gbn_b,
                       const float* __restrict__ fc1_w, const float* __restrict__ fc1_b,
                       const float* __restrict__ fbn_g, const float* __restrict__ fbn_b,
                       const float* __restrict__ fc2_w, const float* __restrict__ fc2_b,
                       float* __restrict__ out){
  __shared__ float z[160];
  __shared__ float zf[128];
  int g = blockIdx.x, t = threadIdx.x;
  if (t < 32){
    float v = gp_b[t];
    for (int j = 0; j < 10; ++j) v += gf[j]*gp_w[j*32 + t];
    v = v*gbn_g[t] + gbn_b[t];
    z[128 + t] = silu_f(v);
  }
  if (t < 128) z[t] = pooled[g*128 + t] / fmaxf(pcnt[g], 1.0f);
  __syncthreads();
  if (t < 128){
    float v = fc1_b[t];
    for (int j = 0; j < 160; ++j) v += z[j]*fc1_w[j*128 + t];
    v = v*fbn_g[t] + fbn_b[t];
    zf[t] = silu_f(v);
  }
  __syncthreads();
  if (t == 0){
    float v = fc2_b[0];
    for (int j = 0; j < 128; ++j) v += zf[j]*fc2_w[j];
    out[g] = v;
  }
}

// ---------------- launcher ----------------
extern "C" void kernel_launch(void* const* d_in, const int* in_sizes, int n_in,
                              void* d_out, int out_size, void* d_ws, size_t ws_size,
                              hipStream_t stream){
  (void)in_sizes; (void)n_in; (void)out_size; (void)ws_size;
  const float* x     = (const float*)d_in[0];
  const int*   ei    = (const int*)  d_in[1];
  const float* ea    = (const float*)d_in[2];
  const int*   batch = (const int*)  d_in[3];
  const float* gf    = (const float*)d_in[4];
  const float* ec1_w1=(const float*)d_in[5],  *ec1_b1=(const float*)d_in[6];
  const float* ec1_w2=(const float*)d_in[7],  *ec1_b2=(const float*)d_in[8];
  const float* ec2_w1=(const float*)d_in[9],  *ec2_b1=(const float*)d_in[10];
  const float* ec2_w2=(const float*)d_in[11], *ec2_b2=(const float*)d_in[12];
  const float* ec3_w1=(const float*)d_in[13], *ec3_b1=(const float*)d_in[14];
  const float* ec3_w2=(const float*)d_in[15], *ec3_b2=(const float*)d_in[16];
  const float* root1 =(const float*)d_in[17], *bias1 =(const float*)d_in[18];
  const float* root2 =(const float*)d_in[19], *bias2 =(const float*)d_in[20];
  const float* root3 =(const float*)d_in[21], *bias3 =(const float*)d_in[22];
  const float* gp_w  =(const float*)d_in[23], *gp_b  =(const float*)d_in[24];
  const float* gbn_g =(const float*)d_in[25], *gbn_b =(const float*)d_in[26];
  const float* fc1_w =(const float*)d_in[27], *fc1_b =(const float*)d_in[28];
  const float* fbn_g =(const float*)d_in[29], *fbn_b =(const float*)d_in[30];
  const float* fc2_w =(const float*)d_in[31], *fc2_b =(const float*)d_in[32];

  const int* src = ei;
  const int* dst = ei + N_EDGES;

  char* w = (char*)d_ws;
  auto alloc = [&](size_t bytes)->char*{ char* p = w; w += (bytes + 255) & ~(size_t)255; return p; };
  __bf16* he   = (__bf16*)alloc((size_t)N_EDGES*128*2);
  __bf16* bv1  = (__bf16*)alloc((size_t)65*4096*2);
  __bf16* bv2  = (__bf16*)alloc((size_t)516*4096*2);
  __bf16* bv3  = (__bf16*)alloc((size_t)516*4096*2);
  __bf16* rbv2 = (__bf16*)alloc((size_t)4*4096*2);
  __bf16* rbv3 = (__bf16*)alloc((size_t)4*4096*2);
  __bf16* xp1  = (__bf16*)alloc((size_t)N_NODES*16*2);
  __bf16* xpA  = (__bf16*)alloc((size_t)N_NODES*128*2);
  __bf16* xpB  = (__bf16*)alloc((size_t)N_NODES*128*2);
  float*  agg  = (float*) alloc((size_t)N_NODES*128*4);
  float*  deg  = (float*) alloc((size_t)N_NODES*4);
  float*  pooled=(float*) alloc((size_t)N_GRAPHS*128*4);
  float*  pcnt = (float*) alloc((size_t)N_GRAPHS*4);

  dim3 B(256), Bm(128);
  // prep
  k_zero<<<(N_NODES+255)/256, B, 0, stream>>>(deg, N_NODES);
  k_deg<<<(N_EDGES+255)/256, B, 0, stream>>>(dst, deg);
  k_recip<<<(N_NODES+255)/256, B, 0, stream>>>(deg);       // deg := 1/max(deg,1)
  k_prepx<<<(N_NODES*16+255)/256, B, 0, stream>>>(x, xp1);
  k_gen_bv<4><<<(65*4096+255)/256,  B, 0, stream>>>(ec1_w2, ec1_b2, bv1, 9,   65);
  k_gen_bv<7><<<(516*4096+255)/256, B, 0, stream>>>(ec2_w2, ec2_b2, bv2, 128, 516);
  k_gen_bv<7><<<(516*4096+255)/256, B, 0, stream>>>(ec3_w2, ec3_b2, bv3, 128, 516);
  k_pack_root<<<(4*4096+255)/256, B, 0, stream>>>(root2, rbv2);
  k_pack_root<<<(4*4096+255)/256, B, 0, stream>>>(root3, rbv3);

  // layer 1 (fin padded 9->16)
  k_edge_hidden<<<(N_EDGES*128+255)/256, B, 0, stream>>>(ea, ec1_w1, ec1_b1, he);
  k_zero<<<(N_NODES*128+255)/256, B, 0, stream>>>(agg, N_NODES*128);
  k_msg_gemm<4><<<N_EDGES/64, Bm, 0, stream>>>(xp1, he, bv1, src, dst, agg, 65);
  k_node1<<<(N_NODES*128+255)/256, B, 0, stream>>>(x, root1, bias1, agg, deg, xpA);

  // layer 2
  k_edge_hidden<<<(N_EDGES*128+255)/256, B, 0, stream>>>(ea, ec2_w1, ec2_b1, he);
  k_zero<<<(N_NODES*128+255)/256, B, 0, stream>>>(agg, N_NODES*128);
  k_msg_gemm<7><<<N_EDGES/64, Bm, 0, stream>>>(xpA, he, bv2, src, dst, agg, 516);
  k_node_wmma<<<N_NODES/32, B, 0, stream>>>(xpA, rbv2, bias2, agg, deg, xpB);

  // layer 3
  k_edge_hidden<<<(N_EDGES*128+255)/256, B, 0, stream>>>(ea, ec3_w1, ec3_b1, he);
  k_zero<<<(N_NODES*128+255)/256, B, 0, stream>>>(agg, N_NODES*128);
  k_msg_gemm<7><<<N_EDGES/64, Bm, 0, stream>>>(xpB, he, bv3, src, dst, agg, 516);
  k_node_wmma<<<N_NODES/32, B, 0, stream>>>(xpB, rbv3, bias3, agg, deg, xpA);

  // pool + head
  k_zero<<<(N_GRAPHS*128+255)/256, B, 0, stream>>>(pooled, N_GRAPHS*128);
  k_zero<<<(N_GRAPHS+255)/256, B, 0, stream>>>(pcnt, N_GRAPHS);
  k_pool<<<(N_NODES*128+255)/256, B, 0, stream>>>(xpA, batch, pooled, pcnt);
  k_head<<<N_GRAPHS, 160, 0, stream>>>(pooled, pcnt, gf, gp_w, gp_b, gbn_g, gbn_b,
                                       fc1_w, fc1_b, fbn_g, fbn_b, fc2_w, fc2_b,
                                       (float*)d_out);
}
